// MultiHeadAttention_82420422410862
// MI455X (gfx1250) — matrix operand order repaired and checked by reading, not compiled, not run on hardware
//
#include <hip/hip_runtime.h>
#include <hip/hip_bf16.h>
#include <cstdint>

// Problem constants
#define S_LEN 2048
#define B_SZ  2
#define D_DIM 1024
#define H_NUM 16
#define DH    64

typedef __attribute__((ext_vector_type(16))) __bf16 v16bf;
typedef __attribute__((ext_vector_type(8)))  float  v8f;
typedef int i32x4 __attribute__((vector_size(16)));

static __device__ inline v8f wmma_bf16(v16bf a, v16bf b, v8f c) {
  // D = A(16x32 bf16) * B(32x16 bf16) + C(16x16 f32)
  return __builtin_amdgcn_wmma_f32_16x16x32_bf16(
      /*neg_a=*/false, a, /*neg_b=*/false, b,
      /*c_mod=*/(short)0, c, /*reuse_a=*/false, /*reuse_b=*/false);
}

// --- gfx1250 async copy global->LDS (ASYNCcnt-tracked) ---------------------
// Builtin prototype (from compiler diagnostic): param0 = i32x4 addrspace(1)*,
// then LDS ptr, imm offset, imm cpol.
#define GADDR(p) ((__attribute__((address_space(1))) i32x4*)(unsigned long long)(uintptr_t)(p))
#define LADDR(p) ((__attribute__((address_space(3))) i32x4*)(unsigned int)(uintptr_t)(p))

static __device__ inline void async_copy_b128(const void* gsrc, void* ldst) {
#if __has_builtin(__builtin_amdgcn_global_load_async_to_lds_b128)
  __builtin_amdgcn_global_load_async_to_lds_b128(GADDR(gsrc), LADDR(ldst), 0, 0);
#else
  asm volatile("global_load_async_to_lds_b128 %0, %1, off"
               :: "v"((unsigned int)(uintptr_t)ldst),
                  "v"((unsigned long long)(uintptr_t)gsrc)
               : "memory");
#endif
}

#if __has_builtin(__builtin_amdgcn_s_wait_asynccnt)
#define WAIT_ASYNC(n) __builtin_amdgcn_s_wait_asynccnt(n)
#else
#define WAIT_ASYNC(n) asm volatile("s_wait_asynccnt %0" :: "i"(n))
#endif

// ---------------------------------------------------------------------------
// Kernel 0: fp32 -> bf16 weight conversion
// ---------------------------------------------------------------------------
__global__ void mha_cvt_bf16(const float* __restrict__ src,
                             __bf16* __restrict__ dst, int n) {
  int i = blockIdx.x * blockDim.x + threadIdx.x;
  if (i < n) dst[i] = (__bf16)src[i];
}

// ---------------------------------------------------------------------------
// Kernel 1/3: GEMM  Out[M,N] = X[M,K] @ W[N,K]^T   (M=4096, N=K=1024)
// MODE 0: X fp32 (query), out bf16 head-major  qh[b][h][s][64]
// MODE 1: X fp32 (key),   out bf16 head-major  kh[b][h][s][64]
// MODE 2: X fp32 (value), out bf16 TRANSPOSED  vt[b][h][d][S]
// MODE 3: X bf16 (attn O),out fp32 plain       out[m][n]
// Workgroup = 64x64 tile, 8 waves; wave = 16(M) x 32(N).
// W tile (64 cols x 32 k) is async-staged into LDS, double-buffered:
// removes the 4x redundant W fetch across M-subtile waves and overlaps the
// next tile's fetch with the current WMMAs (ASYNCcnt pipeline).
// ---------------------------------------------------------------------------
#define WT_STRIDE 40  // padded row stride (elems) to dodge LDS bank conflicts

template <int MODE>
__global__ __launch_bounds__(256) void mha_gemm(const void* __restrict__ Xv,
                                                const __bf16* __restrict__ W,
                                                void* __restrict__ Out) {
  __shared__ __align__(64) __bf16 wtile[2][64 * WT_STRIDE];

  const int w    = threadIdx.x >> 5;
  const int lane = threadIdx.x & 31;
  const int lm   = lane & 15;       // A row / B col / D col within fragment
  const int hi   = lane >> 4;       // half-wave select
  const int kbA  = hi * 16;         // A-fragment K base (contiguous layout)
  const int kbB  = hi * 16;         // B-fragment K base

  const int nb0 = blockIdx.x * 64;                  // workgroup N base
  const int wm0 = blockIdx.y * 64 + (w >> 1) * 16;  // wave M base
  const int wn0 = (w & 1) * 32;                     // wave N base within tile

  const float*  Xf = (const float*)Xv;
  const __bf16* Xb = (const __bf16*)Xv;

  const int arow = wm0 + lm;
  const int cl0  = wn0 + lm;        // local col of fragment b0
  const int cl1  = wn0 + 16 + lm;   // local col of fragment b1

  // async stage: 256 threads x 16B = 64x32 bf16 tile
  auto stage = [&](int buf, int k0) {
    const int col   = threadIdx.x >> 2;
    const int chunk = (threadIdx.x & 3) * 8;
    async_copy_b128(&W[(size_t)(nb0 + col) * D_DIM + k0 + chunk],
                    &wtile[buf][col * WT_STRIDE + chunk]);
  };

  v8f acc0 = {};
  v8f acc1 = {};

  stage(0, 0);
  for (int k0 = 0; k0 < D_DIM; k0 += 32) {
    const int cur = (k0 >> 5) & 1;
    if (k0 + 32 < D_DIM) {
      stage(cur ^ 1, k0 + 32);  // fetch next tile into other buffer
      WAIT_ASYNC(1);            // current tile's copy (per wave) complete
    } else {
      WAIT_ASYNC(0);
    }
    __syncthreads();            // all waves' stage of `cur` visible

    v16bf a;
#pragma unroll
    for (int i = 0; i < 8; i++) {
      if (MODE == 3) {
        a[i]     = Xb[(size_t)arow * D_DIM + k0 + kbA + i];
        a[8 + i] = Xb[(size_t)arow * D_DIM + k0 + kbA + 8 + i];
      } else {
        a[i]     = (__bf16)Xf[(size_t)arow * D_DIM + k0 + kbA + i];
        a[8 + i] = (__bf16)Xf[(size_t)arow * D_DIM + k0 + kbA + 8 + i];
      }
    }
    const __bf16* wl = &wtile[cur][0];
    v16bf b0, b1;
#pragma unroll
    for (int i = 0; i < 16; i++) {
      b0[i] = wl[cl0 * WT_STRIDE + kbB + i];
      b1[i] = wl[cl1 * WT_STRIDE + kbB + i];
    }
    acc0 = wmma_bf16(a, b0, acc0);
    acc1 = wmma_bf16(a, b1, acc1);

    if (k0 + 32 < D_DIM) {  // gfx1250 global_prefetch_b8 for next A chunk
      if (MODE == 3)
        __builtin_prefetch(&Xb[(size_t)arow * D_DIM + k0 + 32 + kbA], 0, 1);
      else
        __builtin_prefetch(&Xf[(size_t)arow * D_DIM + k0 + 32 + kbA], 0, 1);
    }
    __syncthreads();  // done reading `cur` before it is re-staged
  }

  // Epilogue: D layout — lane holds col n=lm, rows m = r + 8*hi
#pragma unroll
  for (int r = 0; r < 8; r++) {
    const int mg = wm0 + r + 8 * hi;
    const int s  = mg >> 1;      // row = s*B + b, B=2
    const int bb = mg & 1;
#pragma unroll
    for (int j = 0; j < 2; j++) {
      const int   ng = nb0 + wn0 + j * 16 + lm;
      const float v  = (j ? acc1[r] : acc0[r]);
      if (MODE == 3) {
        ((float*)Out)[(size_t)mg * D_DIM + ng] = v;
      } else {
        const int h = ng >> 6, d = ng & 63;
        __bf16* O = (__bf16*)Out;
        if (MODE == 2)  // vt[b][h][d][S]
          O[((size_t)(bb * H_NUM + h) * DH + d) * S_LEN + s] = (__bf16)v;
        else            // qh/kh[b][h][s][64]
          O[((size_t)(bb * H_NUM + h) * S_LEN + s) * DH + d] = (__bf16)v;
      }
    }
  }
}

// ---------------------------------------------------------------------------
// Kernel 2: attention core.  One WG per (b, 16-row q-tile), 8 waves.
// LDS: scores 16x2048 f32 (128KB) + coverage 16x2048 f32 (128KB) + O 16x64.
// Per head: QK^T (WMMA) -> exact softmax in LDS -> coverage += P/H ->
//           P*V (WMMA, ds_add_f32 merge) -> O tile to global.
// ---------------------------------------------------------------------------
__global__ __launch_bounds__(256) void mha_attn(
    const __bf16* __restrict__ qh, const __bf16* __restrict__ kh,
    const __bf16* __restrict__ vt, const unsigned char* __restrict__ mask,
    __bf16* __restrict__ oh, float* __restrict__ cov_out) {
  extern __shared__ float smem[];
  float* sc   = smem;                    // [16][2048] scores/probs
  float* cov  = smem + 16 * 2048;        // [16][2048] coverage accumulator
  float* oacc = smem + 2 * 16 * 2048;    // [16][64]   O accumulator

  const int b   = blockIdx.y;
  const int q0  = blockIdx.x * 16;
  const int tid = threadIdx.x;
  const int w   = tid >> 5;
  const int lane = tid & 31;
  const int lm  = lane & 15;
  const int hi  = lane >> 4;
  const int kbA = hi * 16;
  const int kbB = hi * 16;

  for (int i = tid; i < 16 * 2048; i += 256) cov[i] = 0.0f;

  for (int h = 0; h < H_NUM; h++) {
    for (int i = tid; i < 16 * 64; i += 256) oacc[i] = 0.0f;

    // ---- scores: S = (Q K^T) * dh^-0.5, masked -> LDS -------------------
    const __bf16* Qp = qh + ((size_t)(b * H_NUM + h) * S_LEN + q0) * DH;
    v16bf a0, a1;
#pragma unroll
    for (int i = 0; i < 8; i++) {
      a0[i]     = Qp[lm * DH + kbA + i];
      a0[8 + i] = Qp[lm * DH + 8 + kbA + i];
      a1[i]     = Qp[lm * DH + 32 + kbA + i];
      a1[8 + i] = Qp[lm * DH + 40 + kbA + i];
    }
    const __bf16* Kp = kh + (size_t)(b * H_NUM + h) * S_LEN * DH;
    for (int kt = w; kt < S_LEN / 16; kt += 8) {
      const int k0 = kt * 16;
      v16bf b0, b1;
#pragma unroll
      for (int i = 0; i < 16; i++) {
        b0[i] = Kp[(size_t)(k0 + lm) * DH + kbB + i];
        b1[i] = Kp[(size_t)(k0 + lm) * DH + 32 + kbB + i];
      }
      v8f acc = {};
      acc = wmma_bf16(a0, b0, acc);
      acc = wmma_bf16(a1, b1, acc);
#pragma unroll
      for (int r = 0; r < 8; r++) {
        const int m = r + 8 * hi;
        float sv = acc[r] * 0.125f;  // 64^-0.5
        if (mask[((size_t)b * S_LEN + (q0 + m)) * S_LEN + k0 + lm]) sv = -1e30f;
        sc[m * 2048 + k0 + lm] = sv;
      }
    }
    __syncthreads();

    // ---- exact softmax per row (wave handles 2 rows) + coverage ---------
    for (int rr = 0; rr < 2; rr++) {
      const int row = 2 * w + rr;
      float* sr = sc + row * 2048;
      float mx = -3.0e38f;
      for (int k = lane; k < 2048; k += 32) mx = fmaxf(mx, sr[k]);
#pragma unroll
      for (int off = 16; off; off >>= 1) mx = fmaxf(mx, __shfl_xor(mx, off, 32));
      float sum = 0.0f;
      for (int k = lane; k < 2048; k += 32) {
        const float e = __expf(sr[k] - mx);
        sr[k] = e;
        sum += e;
      }
#pragma unroll
      for (int off = 16; off; off >>= 1) sum += __shfl_xor(sum, off, 32);
      const float inv = 1.0f / sum;
      float* cr = cov + row * 2048;
      for (int k = lane; k < 2048; k += 32) {
        const float p = sr[k] * inv;
        sr[k] = p;
        cr[k] += p * (1.0f / H_NUM);
      }
    }
    __syncthreads();

    // ---- O = P * V  (wave: d-tile = w&3, k-half = w>>2) -----------------
    {
      const int nt = w & 3, khalf = w >> 2;
      const int dcol = nt * 16 + lm;
      const __bf16* Vp = vt + ((size_t)(b * H_NUM + h) * DH + dcol) * S_LEN;
      v8f acc = {};
      for (int ks = 0; ks < 32; ks++) {
        const int k0 = khalf * 1024 + ks * 32;
        v16bf a, bb;
#pragma unroll
        for (int i = 0; i < 8; i++) {
          a[i]     = (__bf16)sc[lm * 2048 + k0 + kbA + i];
          a[8 + i] = (__bf16)sc[lm * 2048 + k0 + 8 + kbA + i];
        }
#pragma unroll
        for (int i = 0; i < 16; i++) bb[i] = Vp[k0 + kbB + i];
        acc = wmma_bf16(a, bb, acc);
      }
#pragma unroll
      for (int r = 0; r < 8; r++) {
        const int m = r + 8 * hi;
        atomicAdd(&oacc[m * 64 + dcol], acc[r]);  // ds_add_f32
      }
    }
    __syncthreads();

    // ---- store O tile (bf16, [s][b][h*64+d]) ----------------------------
    for (int i = tid; i < 16 * 64; i += 256) {
      const int m = i >> 6, d = i & 63;
      oh[((size_t)(q0 + m) * B_SZ + b) * D_DIM + h * DH + d] = (__bf16)oacc[i];
    }
    __syncthreads();
  }

  // ---- coverage [B][S][S] -----------------------------------------------
  for (int i = tid; i < 16 * 2048; i += 256) {
    const int m = i >> 11, k = i & 2047;
    cov_out[((size_t)b * S_LEN + q0 + m) * S_LEN + k] = cov[i];
  }
}

// ---------------------------------------------------------------------------
extern "C" void kernel_launch(void* const* d_in, const int* in_sizes, int n_in,
                              void* d_out, int out_size, void* d_ws,
                              size_t ws_size, hipStream_t stream) {
  (void)in_sizes; (void)n_in; (void)out_size; (void)ws_size;

  const float* q_in = (const float*)d_in[0];
  const float* k_in = (const float*)d_in[1];
  const float* v_in = (const float*)d_in[2];
  const unsigned char* mask = (const unsigned char*)d_in[3];
  const float* Wq = (const float*)d_in[4];
  const float* Wk = (const float*)d_in[5];
  const float* Wv = (const float*)d_in[6];
  const float* Wc = (const float*)d_in[7];

  // Workspace layout (bf16 elements): 4 weights (1M each) + 4 act buffers (4M each) = 40MB
  __bf16* wq_b = (__bf16*)d_ws;
  __bf16* wk_b = wq_b + (size_t)D_DIM * D_DIM;
  __bf16* wv_b = wk_b + (size_t)D_DIM * D_DIM;
  __bf16* wc_b = wv_b + (size_t)D_DIM * D_DIM;
  __bf16* qh   = wc_b + (size_t)D_DIM * D_DIM;
  __bf16* kh   = qh + (size_t)S_LEN * B_SZ * D_DIM;
  __bf16* vt   = kh + (size_t)S_LEN * B_SZ * D_DIM;
  __bf16* oh   = vt + (size_t)S_LEN * B_SZ * D_DIM;

  float* out     = (float*)d_out;                       // [S,B,D]
  float* cov_out = out + (size_t)S_LEN * B_SZ * D_DIM;  // [B,S,S]

  const int nW = D_DIM * D_DIM;
  mha_cvt_bf16<<<(nW + 255) / 256, 256, 0, stream>>>(Wq, wq_b, nW);
  mha_cvt_bf16<<<(nW + 255) / 256, 256, 0, stream>>>(Wk, wk_b, nW);
  mha_cvt_bf16<<<(nW + 255) / 256, 256, 0, stream>>>(Wv, wv_b, nW);
  mha_cvt_bf16<<<(nW + 255) / 256, 256, 0, stream>>>(Wc, wc_b, nW);

  dim3 gg(D_DIM / 64, (S_LEN * B_SZ) / 64);  // (16, 64)
  mha_gemm<0><<<gg, 256, 0, stream>>>(q_in, wq_b, qh);
  mha_gemm<1><<<gg, 256, 0, stream>>>(k_in, wk_b, kh);
  mha_gemm<2><<<gg, 256, 0, stream>>>(v_in, wv_b, vt);

  const size_t lds_bytes = (2 * 16 * 2048 + 16 * 64) * sizeof(float);  // 260KB
  dim3 ga(S_LEN / 16, B_SZ);  // (128, 2)
  mha_attn<<<ga, 256, lds_bytes, stream>>>(qh, kh, vt, mask, oh, cov_out);

  mha_gemm<3><<<gg, 256, 0, stream>>>(oh, wc_b, out);
}
